// MoME_SAGE_39152921870441
// MI455X (gfx1250) — matrix operand order repaired and checked by reading, not compiled, not run on hardware
//
#include <hip/hip_runtime.h>
#include <hip/hip_bf16.h>

// ---------------- types ----------------
typedef __attribute__((ext_vector_type(16))) __bf16 v16bf;
typedef __attribute__((ext_vector_type(8)))  __bf16 v8bf;
typedef __attribute__((ext_vector_type(8)))  float  v8f;

#define BB   16
#define NN   1024
#define DD   768
#define HH   12
#define DHH  64
#define SSX  32
#define DFF  3072
#define ROWS (BB*NN)          // 16384 rows per modality
#define CHUNK 4096            // FFN row chunk

struct ModParams {
  const float *ctx_ln_g, *ctx_ln_b, *ctx_W, *ctx_b;
  const float *gen_k, *gen_v, *alpha;
  const float *ln_g, *ln_b, *Wout;
  const float *ffn_ln_g, *ffn_ln_b;
  const float *W1, *b1, *W2, *b2;
};
struct AllParams { ModParams mod[3]; const float *MkX, *MvX; };

__device__ __forceinline__ float gelu_exact(float x) {
  return 0.5f * x * (1.0f + erff(x * 0.70710678118654752f));
}

// ---------------- pooling: t,a,v = mean over N ----------------
__global__ void pool_kernel(const float* __restrict__ H0, const float* __restrict__ H1,
                            const float* __restrict__ H2, float* __restrict__ pooled) {
  int idx = blockIdx.x * 256 + threadIdx.x;          // 3*16*768 = 36864
  if (idx >= 3 * BB * DD) return;
  int mod = idx / (BB * DD); int rem = idx % (BB * DD);
  int b = rem / DD, d = rem % DD;
  const float* Hp = (mod == 0) ? H0 : ((mod == 1) ? H1 : H2);
  const float* p = Hp + ((size_t)b * NN) * DD + d;
  float s = 0.f;
  for (int n = 0; n < NN; n++) s += p[(size_t)n * DD];
  pooled[idx] = s * (1.f / (float)NN);
}

// ---------------- ctx: cX = gelu(LN(concat) @ ctx_W + ctx_b) ----------------
__global__ void ctx_kernel(const float* __restrict__ pooled, AllParams P, float* __restrict__ cX) {
  __shared__ float row[2 * DD];
  __shared__ float red[256];
  const int mod = blockIdx.x >> 4, b = blockIdx.x & 15;
  // cT = ctx(a,v); cA = ctx(t,v); cV = ctx(t,a)
  const int fi = (mod == 0) ? 1 : 0;
  const int si = (mod == 2) ? 1 : 2;
  const float* p1 = pooled + (fi * BB + b) * DD;
  const float* p2 = pooled + (si * BB + b) * DD;
  for (int i = threadIdx.x; i < DD; i += 256) { row[i] = p1[i]; row[DD + i] = p2[i]; }
  __syncthreads();
  float loc = 0.f;
  for (int i = threadIdx.x; i < 2 * DD; i += 256) loc += row[i];
  red[threadIdx.x] = loc; __syncthreads();
  for (int st = 128; st > 0; st >>= 1) { if (threadIdx.x < st) red[threadIdx.x] += red[threadIdx.x + st]; __syncthreads(); }
  float mean = red[0] * (1.f / (2.f * DD)); __syncthreads();
  loc = 0.f;
  for (int i = threadIdx.x; i < 2 * DD; i += 256) { float d = row[i] - mean; loc += d * d; }
  red[threadIdx.x] = loc; __syncthreads();
  for (int st = 128; st > 0; st >>= 1) { if (threadIdx.x < st) red[threadIdx.x] += red[threadIdx.x + st]; __syncthreads(); }
  float rstd = rsqrtf(red[0] * (1.f / (2.f * DD)) + 1e-5f);
  const ModParams q = P.mod[mod];
  __syncthreads();
  for (int i = threadIdx.x; i < 2 * DD; i += 256)
    row[i] = (row[i] - mean) * rstd * q.ctx_ln_g[i] + q.ctx_ln_b[i];
  __syncthreads();
  for (int j = threadIdx.x; j < DD; j += 256) {
    float acc = q.ctx_b[j];
    for (int k = 0; k < 2 * DD; k++) acc += row[k] * q.ctx_W[(size_t)k * DD + j];
    cX[(mod * BB + b) * DD + j] = gelu_exact(acc);
  }
}

// ---------------- cbar = mean over batch of cX ----------------
__global__ void cbar_kernel(const float* __restrict__ cX, float* __restrict__ cbar) {
  int idx = blockIdx.x * 256 + threadIdx.x;   // 3*768
  if (idx >= 3 * DD) return;
  int mod = idx / DD, d = idx % DD;
  float s = 0.f;
  for (int b = 0; b < BB; b++) s += cX[(mod * BB + b) * DD + d];
  cbar[idx] = s * (1.f / (float)BB);
}

// ---------------- Mk/Mv = a*MX + (1-a)*(cbar @ gen) ----------------
__global__ void kvgen_kernel(const float* __restrict__ cbar, AllParams P,
                             float* __restrict__ Mk, float* __restrict__ Mv) {
  int idx = blockIdx.x * 256 + threadIdx.x;   // 3*24576
  if (idx >= 3 * HH * SSX * DHH) return;
  int mod = idx / (HH * SSX * DHH), j = idx % (HH * SSX * DHH);
  const ModParams q = P.mod[mod];
  const float* cb = cbar + mod * DD;
  float ak = 0.f, av = 0.f;
  for (int d = 0; d < DD; d++) {
    float c = cb[d];
    ak += c * q.gen_k[(size_t)d * (HH * SSX * DHH) + j];
    av += c * q.gen_v[(size_t)d * (HH * SSX * DHH) + j];
  }
  float a = fminf(fmaxf(q.alpha[0], 0.f), 1.f);
  Mk[idx] = a * P.MkX[j] + (1.f - a) * ak;
  Mv[idx] = a * P.MvX[j] + (1.f - a) * av;
}

// ---------------- row LN -> bf16 (used for xhat and for ffn hln) ----------------
__global__ void ln_to_bf16_kernel(const float* __restrict__ X, const float* __restrict__ g,
                                  const float* __restrict__ bb, __bf16* __restrict__ Y) {
  __shared__ float red[256];
  const int row = blockIdx.x;
  const float* x = X + (size_t)row * DD;
  float vals[3]; float v0 = 0.f;
#pragma unroll
  for (int i = 0; i < 3; i++) { vals[i] = x[threadIdx.x + 256 * i]; v0 += vals[i]; }
  red[threadIdx.x] = v0; __syncthreads();
  for (int s = 128; s > 0; s >>= 1) { if (threadIdx.x < s) red[threadIdx.x] += red[threadIdx.x + s]; __syncthreads(); }
  float mean = red[0] * (1.f / (float)DD); __syncthreads();
  float v1 = 0.f;
#pragma unroll
  for (int i = 0; i < 3; i++) { float d = vals[i] - mean; v1 += d * d; }
  red[threadIdx.x] = v1; __syncthreads();
  for (int s = 128; s > 0; s >>= 1) { if (threadIdx.x < s) red[threadIdx.x] += red[threadIdx.x + s]; __syncthreads(); }
  float rstd = rsqrtf(red[0] * (1.f / (float)DD) + 1e-5f);
#pragma unroll
  for (int i = 0; i < 3; i++) {
    int c = threadIdx.x + 256 * i;
    Y[(size_t)row * DD + c] = (__bf16)((vals[i] - mean) * rstd * g[c] + bb[c]);
  }
}

// ---------------- scores: R[b,h,n,s] = sum_d xhat[b,n,h*64+d]*Mk[h,s,d] ----------------
__global__ void scores_kernel(const __bf16* __restrict__ xhat, const float* __restrict__ Mk,
                              float* __restrict__ R) {
  __shared__ float mk[SSX * 65];
  const int blk = blockIdx.x;             // bh*16 + nt
  const int nt = blk & 15, bh = blk >> 4;
  const int h = bh % HH, b = bh / HH;
  for (int i = threadIdx.x; i < SSX * DHH; i += 256)
    mk[(i >> 6) * 65 + (i & 63)] = Mk[h * (SSX * DHH) + i];
  __syncthreads();
  const int n = nt * 64 + (threadIdx.x >> 2);
  const int soff = (threadIdx.x & 3) * 8;
  const __bf16* xr = xhat + ((size_t)b * NN + n) * DD + h * DHH;
  float x[DHH];
#pragma unroll
  for (int d = 0; d < DHH; d++) x[d] = (float)xr[d];
  float accs[8];
#pragma unroll
  for (int si = 0; si < 8; si++) accs[si] = 0.f;
  for (int d = 0; d < DHH; d++) {
    float xv = x[d];
#pragma unroll
    for (int si = 0; si < 8; si++) accs[si] += xv * mk[(soff + si) * 65 + d];
  }
  float* Rp = R + ((size_t)bh * NN + n) * SSX + soff;
#pragma unroll
  for (int si = 0; si < 8; si++) Rp[si] = accs[si];
}

// ---------------- column softmax stats over N per (b,h,s) ----------------
__global__ void colreduce_kernel(const float* __restrict__ R, float* __restrict__ colmax,
                                 float* __restrict__ colsum) {
  __shared__ float red[256];
  const int bh = blockIdx.x;
  const int s = threadIdx.x & 31, g = threadIdx.x >> 5;   // 8 groups x 32 slots
  const float* Rp = R + (size_t)bh * NN * SSX;
  float m = -3.4e38f;
  for (int n = g; n < NN; n += 8) m = fmaxf(m, Rp[n * SSX + s]);
  red[threadIdx.x] = m; __syncthreads();
  if (g == 0) { float mm = red[s]; for (int i = 1; i < 8; i++) mm = fmaxf(mm, red[i * 32 + s]); red[s] = mm; }
  __syncthreads();
  float cm = red[s];
  __syncthreads();
  float sum = 0.f;
  for (int n = g; n < NN; n += 8) sum += expf(Rp[n * SSX + s] - cm);
  red[threadIdx.x] = sum; __syncthreads();
  if (g == 0) {
    float ss = red[s]; for (int i = 1; i < 8; i++) ss += red[i * 32 + s];
    colmax[bh * SSX + s] = cm; colsum[bh * SSX + s] = ss;
  }
}

// ---------------- normalize + O = R_norm @ Mv (bf16 out) ----------------
__global__ void normO_kernel(const float* __restrict__ R, const float* __restrict__ colmax,
                             const float* __restrict__ colsum, const float* __restrict__ Mv,
                             __bf16* __restrict__ O) {
  __shared__ float mv[SSX * DHH];
  __shared__ float e[64 * 33];
  __shared__ float rinv[64];
  __shared__ float cmx[SSX], csm[SSX];
  const int blk = blockIdx.x;
  const int nt = blk & 15, bh = blk >> 4;
  const int h = bh % HH, b = bh / HH;
  for (int i = threadIdx.x; i < SSX * DHH; i += 256) mv[i] = Mv[h * (SSX * DHH) + i];
  if (threadIdx.x < SSX) { cmx[threadIdx.x] = colmax[bh * SSX + threadIdx.x]; csm[threadIdx.x] = colsum[bh * SSX + threadIdx.x]; }
  __syncthreads();
  const float* Rp = R + ((size_t)bh * NN + nt * 64) * SSX;
  for (int i = threadIdx.x; i < 64 * SSX; i += 256) {
    int r = i >> 5, s = i & 31;
    e[r * 33 + s] = expf(Rp[r * SSX + s] - cmx[s]) / csm[s];
  }
  __syncthreads();
  if (threadIdx.x < 64) {
    float ss = 0.f;
    for (int s = 0; s < SSX; s++) ss += e[threadIdx.x * 33 + s];
    rinv[threadIdx.x] = 1.f / (1e-9f + ss);
  }
  __syncthreads();
  for (int i = threadIdx.x; i < 64 * DHH; i += 256) {
    int r = i >> 6, d = i & 63;
    float acc = 0.f;
#pragma unroll
    for (int s = 0; s < SSX; s++) acc += e[r * 33 + s] * mv[s * DHH + d];
    O[((size_t)b * NN + nt * 64 + r) * DD + h * DHH + d] = (__bf16)(acc * rinv[r]);
  }
}

// ---------------- f32 -> bf16 transpose convert: dst(N x K) = src(K x N)^T ----------------
__global__ void f32_to_bf16_T_kernel(const float* __restrict__ src, __bf16* __restrict__ dst,
                                     int K, int N) {
  __shared__ float tile[32][33];
  const int kt = blockIdx.y * 32, nt = blockIdx.x * 32;
  const int tx = threadIdx.x & 31, ty = threadIdx.x >> 5;   // 32x8
#pragma unroll
  for (int i = 0; i < 32; i += 8)
    tile[ty + i][tx] = src[(size_t)(kt + ty + i) * N + nt + tx];
  __syncthreads();
#pragma unroll
  for (int i = 0; i < 32; i += 8)
    dst[(size_t)(nt + ty + i) * K + kt + tx] = (__bf16)tile[tx][ty + i];
}

// ---------------- WMMA bf16 GEMM: C(MxN) = A(MxK) x Bt(NxK)^T, fused epilogues --------
// Bt is the PRE-TRANSPOSED weight (N x K row-major) -> both tiles stage with b128 ops.
// Double-buffered LDS: one barrier per k-step; next tile's global loads overlap WMMAs.
// mode 0: outF = resid + acc                    (y = Fx + O@Wout)
// mode 1: outB = bf16(gelu(acc + bias))         (mid = gelu(h@W1+b1))
// mode 2: outF = outF + acc + bias              (out = y + mid@W2 + b2, in-place)
__global__ __launch_bounds__(256) void gemm_bf16_kernel(
    const __bf16* __restrict__ A, const __bf16* __restrict__ Bt,
    int M, int N, int K,
    const float* __restrict__ bias, const float* __restrict__ resid,
    float* __restrict__ outF, __bf16* __restrict__ outB, int mode) {
  __shared__ __bf16 As[2][128][40];
  __shared__ __bf16 Bs[2][128][40];
  const int tid = threadIdx.x;
  const int lane = tid & 31, wid = tid >> 5;    // wave32: 8 waves
  const int m0 = blockIdx.y * 128, n0 = blockIdx.x * 128;
  const int wm = (wid & 3) * 32;                // wave m offset (4 waves in M)
  const int wn = (wid >> 2) * 64;               // wave n offset (2 waves in N)
  const int row = tid >> 1, colh = (tid & 1) * 16;   // 128 rows x two 16-elt halves
  const int kb  = (lane < 16) ? 0 : 8;    // A fragment k-base (16-bit WMMA layout)
  const int kb2 = (lane < 16) ? 0 : 8;    // B fragment k-base
  const int ml  = lane & 15;
  v8f acc[2][4] = {};

  const __bf16* ag = A  + (size_t)(m0 + row) * K + colh;
  const __bf16* bg = Bt + (size_t)(n0 + row) * K + colh;

  // stage first tile into buffer 0
  *(v8bf*)&As[0][row][colh]     = *(const v8bf*)(ag);
  *(v8bf*)&As[0][row][colh + 8] = *(const v8bf*)(ag + 8);
  *(v8bf*)&Bs[0][row][colh]     = *(const v8bf*)(bg);
  *(v8bf*)&Bs[0][row][colh + 8] = *(const v8bf*)(bg + 8);

  const int steps = K >> 5;
  for (int s = 0; s < steps; s++) {
    const int cur = s & 1, nxt = cur ^ 1;
    __syncthreads();
    if (s + 1 < steps) {   // stage next tile into other buffer while computing this one
      const __bf16* agn = ag + (size_t)(s + 1) * 32;
      const __bf16* bgn = bg + (size_t)(s + 1) * 32;
      *(v8bf*)&As[nxt][row][colh]     = *(const v8bf*)(agn);
      *(v8bf*)&As[nxt][row][colh + 8] = *(const v8bf*)(agn + 8);
      *(v8bf*)&Bs[nxt][row][colh]     = *(const v8bf*)(bgn);
      *(v8bf*)&Bs[nxt][row][colh + 8] = *(const v8bf*)(bgn + 8);
      if (s + 2 < steps) {             // -> global_prefetch_b8
        __builtin_prefetch(agn + 32, 0, 0);
        __builtin_prefetch(bgn + 32, 0, 0);
      }
    }
    v16bf afrag[2], bfrag[4];
#pragma unroll
    for (int i = 0; i < 2; i++) {
      const __bf16* ap = &As[cur][wm + i * 16 + ml][0];
      v8bf lo = *(const v8bf*)(ap + kb);
      v8bf hi = *(const v8bf*)(ap + kb + 16);
#pragma unroll
      for (int e = 0; e < 8; e++) { afrag[i][e] = lo[e]; afrag[i][e + 8] = hi[e]; }
    }
#pragma unroll
    for (int j = 0; j < 4; j++) {
      const __bf16* bp = &Bs[cur][wn + j * 16 + ml][0];
      v8bf lo = *(const v8bf*)(bp + kb2);
      v8bf hi = *(const v8bf*)(bp + kb2 + 16);
#pragma unroll
      for (int e = 0; e < 8; e++) { bfrag[j][e] = lo[e]; bfrag[j][e + 8] = hi[e]; }
    }
#pragma unroll
    for (int i = 0; i < 2; i++)
#pragma unroll
      for (int j = 0; j < 4; j++)
        acc[i][j] = __builtin_amdgcn_wmma_f32_16x16x32_bf16(
            false, afrag[i], false, bfrag[j], (short)0, acc[i][j], false, false);
  }

  // epilogue: C 16x16 f32 layout: m = mbase + (lane>=16)*8 + r, n = nbase + (lane&15)
#pragma unroll
  for (int i = 0; i < 2; i++) {
#pragma unroll
    for (int j = 0; j < 4; j++) {
      const int mb = m0 + wm + i * 16 + (lane >> 4) * 8;
      const int nb = n0 + wn + j * 16 + ml;
      const float bv = (mode != 0) ? bias[nb] : 0.f;
#pragma unroll
      for (int r = 0; r < 8; r++) {
        const size_t off = (size_t)(mb + r) * N + nb;
        const float v = acc[i][j][r];
        if (mode == 0)      outF[off] = resid[off] + v;
        else if (mode == 1) outB[off] = (__bf16)gelu_exact(v + bv);
        else                outF[off] = outF[off] + v + bv;
      }
    }
  }
}

// ---------------- host side ----------------
static void build_params(void* const* d_in, const int* in_sizes, int n_in,
                         const float* HX[3], AllParams& P) {
  const bool sorted = (n_in >= 53) && (in_sizes[3] == DD * DFF);   // W1 first => tree_flatten sorted keys
  if (!sorted) {
    // insertion order: HT,HA,HV, T-module(16), A-module(16), V-module(16), MkX, MvX
    HX[0] = (const float*)d_in[0]; HX[1] = (const float*)d_in[1]; HX[2] = (const float*)d_in[2];
    for (int m = 0; m < 3; m++) {
      const int base = 3 + m * 16;
      ModParams& q = P.mod[m];
      q.ctx_ln_g = (const float*)d_in[base + 0]; q.ctx_ln_b = (const float*)d_in[base + 1];
      q.ctx_W    = (const float*)d_in[base + 2]; q.ctx_b    = (const float*)d_in[base + 3];
      q.gen_k    = (const float*)d_in[base + 4]; q.gen_v    = (const float*)d_in[base + 5];
      q.alpha    = (const float*)d_in[base + 6];
      q.ln_g     = (const float*)d_in[base + 7]; q.ln_b     = (const float*)d_in[base + 8];
      q.Wout     = (const float*)d_in[base + 9];
      q.ffn_ln_g = (const float*)d_in[base + 10]; q.ffn_ln_b = (const float*)d_in[base + 11];
      q.W1 = (const float*)d_in[base + 12]; q.b1 = (const float*)d_in[base + 13];
      q.W2 = (const float*)d_in[base + 14]; q.b2 = (const float*)d_in[base + 15];
    }
    P.MkX = (const float*)d_in[51]; P.MvX = (const float*)d_in[52];
  } else {
    // sorted flatten: HA,HT,HV, A(3..18), MkX(19), MvX(20), T(21..36), V(37..52)
    HX[0] = (const float*)d_in[1]; HX[1] = (const float*)d_in[0]; HX[2] = (const float*)d_in[2];
    const int mbase[3] = {21, 3, 37};   // T, A, V
    for (int m = 0; m < 3; m++) {
      const int base = mbase[m];
      ModParams& q = P.mod[m];
      // keys sorted: W1,W2,Wout,alpha,b1,b2,ctx_W,ctx_b,ctx_ln_b,ctx_ln_g,ffn_ln_b,ffn_ln_g,gen_k,gen_v,ln_b,ln_g
      q.W1 = (const float*)d_in[base + 0]; q.W2 = (const float*)d_in[base + 1]; q.Wout = (const float*)d_in[base + 2];
      q.alpha = (const float*)d_in[base + 3]; q.b1 = (const float*)d_in[base + 4]; q.b2 = (const float*)d_in[base + 5];
      q.ctx_W = (const float*)d_in[base + 6]; q.ctx_b = (const float*)d_in[base + 7];
      q.ctx_ln_b = (const float*)d_in[base + 8]; q.ctx_ln_g = (const float*)d_in[base + 9];
      q.ffn_ln_b = (const float*)d_in[base + 10]; q.ffn_ln_g = (const float*)d_in[base + 11];
      q.gen_k = (const float*)d_in[base + 12]; q.gen_v = (const float*)d_in[base + 13];
      q.ln_b = (const float*)d_in[base + 14]; q.ln_g = (const float*)d_in[base + 15];
    }
    P.MkX = (const float*)d_in[19]; P.MvX = (const float*)d_in[20];
  }
}

extern "C" void kernel_launch(void* const* d_in, const int* in_sizes, int n_in,
                              void* d_out, int out_size, void* d_ws, size_t ws_size,
                              hipStream_t stream) {
  (void)out_size; (void)ws_size;
  const float* HX[3]; AllParams P;
  build_params(d_in, in_sizes, n_in, HX, P);

  char* w = (char*)d_ws;
  auto alloc = [&](size_t bytes) { void* p = (void*)w; w += (bytes + 255) & ~(size_t)255; return p; };
  float*  pooled = (float*)alloc((size_t)3 * BB * DD * 4);
  float*  cX     = (float*)alloc((size_t)3 * BB * DD * 4);
  float*  cbar   = (float*)alloc((size_t)3 * DD * 4);
  float*  Mk     = (float*)alloc((size_t)3 * HH * SSX * DHH * 4);
  float*  Mv     = (float*)alloc((size_t)3 * HH * SSX * DHH * 4);
  float*  colmax = (float*)alloc((size_t)BB * HH * SSX * 4);
  float*  colsum = (float*)alloc((size_t)BB * HH * SSX * 4);
  __bf16* WoutT  = (__bf16*)alloc((size_t)DD * DD * 2);     // transposed bf16 weights
  __bf16* W1T    = (__bf16*)alloc((size_t)DD * DFF * 2);
  __bf16* W2T    = (__bf16*)alloc((size_t)DFF * DD * 2);
  __bf16* xhatB  = (__bf16*)alloc((size_t)ROWS * DD * 2);
  __bf16* OB     = (__bf16*)alloc((size_t)ROWS * DD * 2);
  __bf16* midB   = (__bf16*)alloc((size_t)CHUNK * DFF * 2);
  float*  R      = (float*)alloc((size_t)BB * HH * NN * SSX * 4);

  pool_kernel<<<(3 * BB * DD + 255) / 256, 256, 0, stream>>>(HX[0], HX[1], HX[2], pooled);
  ctx_kernel<<<48, 256, 0, stream>>>(pooled, P, cX);
  cbar_kernel<<<(3 * DD + 255) / 256, 256, 0, stream>>>(cX, cbar);
  kvgen_kernel<<<(3 * HH * SSX * DHH + 255) / 256, 256, 0, stream>>>(cbar, P, Mk, Mv);

  for (int m = 0; m < 3; m++) {
    const ModParams& q = P.mod[m];
    float* outSlice = (float*)d_out + (size_t)m * ROWS * DD;
    { dim3 g(DD / 32, DD / 32);
      f32_to_bf16_T_kernel<<<g, 256, 0, stream>>>(q.Wout, WoutT, DD, DD); }
    { dim3 g(DFF / 32, DD / 32);
      f32_to_bf16_T_kernel<<<g, 256, 0, stream>>>(q.W1, W1T, DD, DFF); }
    { dim3 g(DD / 32, DFF / 32);
      f32_to_bf16_T_kernel<<<g, 256, 0, stream>>>(q.W2, W2T, DFF, DD); }

    ln_to_bf16_kernel<<<ROWS, 256, 0, stream>>>(HX[m], q.ln_g, q.ln_b, xhatB);
    scores_kernel<<<BB * HH * (NN / 64), 256, 0, stream>>>(xhatB, Mk + m * HH * SSX * DHH, R);
    colreduce_kernel<<<BB * HH, 256, 0, stream>>>(R, colmax, colsum);
    normO_kernel<<<BB * HH * (NN / 64), 256, 0, stream>>>(R, colmax, colsum, Mv + m * HH * SSX * DHH, OB);

    { // y = Fx + O @ Wout   (store into d_out slice)
      dim3 g(DD / 128, ROWS / 128);
      gemm_bf16_kernel<<<g, 256, 0, stream>>>(OB, WoutT, ROWS, DD, DD,
                                              nullptr, HX[m], outSlice, nullptr, 0);
    }
    ln_to_bf16_kernel<<<ROWS, 256, 0, stream>>>(outSlice, q.ffn_ln_g, q.ffn_ln_b, xhatB);
    for (int c = 0; c < ROWS / CHUNK; c++) {
      dim3 g1(DFF / 128, CHUNK / 128);
      gemm_bf16_kernel<<<g1, 256, 0, stream>>>(xhatB + (size_t)c * CHUNK * DD, W1T,
                                               CHUNK, DFF, DD, q.b1, nullptr, nullptr, midB, 1);
      dim3 g2(DD / 128, CHUNK / 128);
      gemm_bf16_kernel<<<g2, 256, 0, stream>>>(midB, W2T, CHUNK, DD, DFF,
                                               q.b2, nullptr, outSlice + (size_t)c * CHUNK * DD, nullptr, 2);
    }
  }
}
